// LSTM_45337674776685
// MI455X (gfx1250) — matrix operand order repaired and checked by reading, not compiled, not run on hardware
//
#include <hip/hip_runtime.h>

// ---------------------------------------------------------------------------
// Types for CDNA5 WMMA (wave32): D(16x16 f32) = A(16x32 bf16) x B(32x16 bf16) + C
// ---------------------------------------------------------------------------
typedef __bf16 bf16_t;
typedef __attribute__((ext_vector_type(16))) __bf16 bf16x16;
typedef __attribute__((ext_vector_type(8)))  float  floatx8;

#define DEV_INLINE __device__ __forceinline__

union FragCast { int4 v[2]; bf16x16 f; };

// A fragment: rows M = lane&15 ; lanes<16 hold K = {0..7,16..23}+k0, lanes>=16 hold +8.
// Both halves are contiguous 8-element runs -> two b128 loads.
DEV_INLINE bf16x16 frag_a(const bf16_t* base, int ld, int k0, int lane) {
  const int m  = lane & 15;
  const int kb = (lane & 16) ? 8 : 0;
  const bf16_t* p = base + (size_t)m * ld + k0 + kb;
  FragCast u;
  u.v[0] = *(const int4*)(p);
  u.v[1] = *(const int4*)(p + 16);
  return u.f;
}

// B fragment from row-major W[N][K] (i.e. computes W^T):
// cols N = lane&15 ; lanes<16 hold K = k0+{0..7,16..23}, lanes>=16 hold +8.
DEV_INLINE bf16x16 frag_b(const bf16_t* W, int ld, int n0, int k0, int lane) {
  const int n  = lane & 15;
  const int ko = (lane & 16) ? 8 : 0;
  const bf16_t* p = W + (size_t)(n0 + n) * ld + k0 + ko;
  FragCast u;
  u.v[0] = *(const int4*)(p);
  u.v[1] = *(const int4*)(p + 16);
  return u.f;
}

DEV_INLINE floatx8 wmma_bf16(bf16x16 a, bf16x16 b, floatx8 c) {
  return __builtin_amdgcn_wmma_f32_16x16x32_bf16(
      /*neg_a=*/false, a, /*neg_b=*/false, b,
      /*c_mod=*/(short)0, c, /*reuse_a=*/false, /*reuse_b=*/false);
}

DEV_INLINE floatx8 bcast8(float v) {
  floatx8 c;
#pragma unroll
  for (int r = 0; r < 8; ++r) c[r] = v;
  return c;
}

DEV_INLINE float sigmoidf_(float x) { return 1.0f / (1.0f + expf(-x)); }
DEV_INLINE float siluf_(float x)    { return x / (1.0f + expf(-x)); }

// ---------------------------------------------------------------------------
// CDNA5 async global->LDS copy (GLOBAL_LOAD_ASYNC_TO_LDS_B128, ASYNCcnt).
// GV mode: VDST = LDS byte offset (VGPR), VADDR = 64-bit global address.
// Kernels using this have only dynamic LDS, so (ptr - smem) is the LDS offset.
// ---------------------------------------------------------------------------
DEV_INLINE void async_g2l_b128(const char* smem, const void* lds_dst,
                               const void* gsrc) {
  unsigned lds_off = (unsigned)((const char*)lds_dst - smem);
  asm volatile("global_load_async_to_lds_b128 %0, %1, off"
               :: "v"(lds_off), "v"(gsrc) : "memory");
}
DEV_INLINE void wait_async0() {
  asm volatile("s_wait_asynccnt 0" ::: "memory");
}

// ---------------------------------------------------------------------------
// Prep kernels
// ---------------------------------------------------------------------------
__global__ void k_cvt_bf16(const float* __restrict__ s, bf16_t* __restrict__ d, int n) {
  int i = blockIdx.x * 256 + threadIdx.x;
  if (i < n) d[i] = (bf16_t)s[i];
}

__global__ void k_bias_sum(const float* __restrict__ a, const float* __restrict__ b,
                           float* __restrict__ o, int n) {
  int i = blockIdx.x * 256 + threadIdx.x;
  if (i < n) o[i] = a[i] + b[i];
}

// x: (B=32, F=64, S=2048) f32 -> x_t: (B, S, F) bf16
__global__ void k_xt(const float* __restrict__ x, bf16_t* __restrict__ xt) {
  int i = blockIdx.x * 256 + threadIdx.x;  // B*S*F = 4194304
  if (i < 32 * 2048 * 64) {
    int f = i & 63;
    int s = (i >> 6) & 2047;
    int b = i >> 17;
    xt[i] = (bf16_t)x[((size_t)b * 64 + f) * 2048 + s];
  }
}

__global__ void k_zero_u32(unsigned* __restrict__ p, int n) {
  int i = blockIdx.x * 256 + threadIdx.x;
  if (i < n) p[i] = 0u;
}

// ---------------------------------------------------------------------------
// Persistent LSTM layer kernel.
// Grid = 8 WGs: blockIdx = bt*4 + hsl ; bt in {0,1} (16 batch rows), hsl in 0..3
// (64 hidden units -> 256 gate rows i/f/g/o). W_hh / W_ih slices live in LDS for
// all 2048 steps (async-loaded once). Per-step h/x staging uses async LDS loads.
// Per-step sync among the 4 WGs of a batch tile via monotonic L2 atomic counter
// (release/acquire, agent scope) + s_sleep spin. h double-buffered in global.
// ---------------------------------------------------------------------------
template <int KIN>
__global__ void lstm_layer_kernel(const bf16_t* __restrict__ xin,   // (B,S,KIN)
                                  const bf16_t* __restrict__ Wih,   // (4H,KIN)
                                  const bf16_t* __restrict__ Whh,   // (4H,H)
                                  const float*  __restrict__ bias,  // (4H) b_ih+b_hh
                                  bf16_t* __restrict__ hout,        // (B,S,H)
                                  bf16_t* __restrict__ hbuf,        // (2,B,H), zeroed
                                  unsigned* __restrict__ bar) {     // counters, zeroed
  constexpr int H = 256, S = 2048;
  extern __shared__ char smem[];
  bf16_t* whh_l = (bf16_t*)smem;            // 256 x 256
  bf16_t* wih_l = whh_l + 256 * 256;        // 256 x KIN
  bf16_t* hs    = wih_l + 256 * KIN;        // 16 x 256
  bf16_t* xs    = hs + 16 * 256;            // 16 x KIN (cap 16x256)
  float*  gl    = (float*)(xs + 16 * 256);  // 16 x 256 gate pre-activations
  float*  cl    = gl + 16 * 256;            // 16 x 64 cell state (persistent)

  const int tid  = threadIdx.x;
  const int lane = tid & 31, wv = tid >> 5;
  const int bt   = blockIdx.x >> 2, hsl = blockIdx.x & 3;

  // Async-load weight slices into LDS (row per thread: n_local = tid = g*64+jj).
  {
    const int g = tid >> 6, jj = tid & 63;
    const int grow = g * 256 + hsl * 64 + jj;
#pragma unroll
    for (int i = 0; i < H / 8; ++i)
      async_g2l_b128(smem, whh_l + (size_t)tid * H + i * 8,
                     Whh + (size_t)grow * H + i * 8);
#pragma unroll
    for (int i = 0; i < KIN / 8; ++i)
      async_g2l_b128(smem, wih_l + (size_t)tid * KIN + i * 8,
                     Wih + (size_t)grow * KIN + i * 8);
  }
#pragma unroll
  for (int i = 0; i < 4; ++i) cl[tid * 4 + i] = 0.0f;

  const int n0 = wv * 32;  // wave owns gate cols [n0, n0+32)
  float bv0, bv1;
  {
    int nl0 = n0 + (lane & 15), nl1 = nl0 + 16;
    bv0 = bias[(nl0 >> 6) * 256 + hsl * 64 + (nl0 & 63)];
    bv1 = bias[(nl1 >> 6) * 256 + hsl * 64 + (nl1 & 63)];
  }
  wait_async0();
  __syncthreads();

  for (int t = 0; t < S; ++t) {
    // Async-stage h_prev (16x256) and x_t (16xKIN) into LDS.
    {
      const bf16_t* hsrc = hbuf + (size_t)(t & 1) * (32 * H) + (size_t)bt * 16 * H;
      for (int i = tid; i < 16 * H / 8; i += 256)
        async_g2l_b128(smem, hs + i * 8, hsrc + i * 8);
    }
    for (int i = tid; i < 16 * (KIN / 8); i += 256) {
      int m = i / (KIN / 8), c = i % (KIN / 8);
      async_g2l_b128(smem, xs + (size_t)m * KIN + c * 8,
                     xin + ((size_t)(bt * 16 + m) * S + t) * KIN + c * 8);
    }
    wait_async0();
    __syncthreads();

    floatx8 acc0 = bcast8(bv0), acc1 = bcast8(bv1);
#pragma unroll
    for (int kt = 0; kt < KIN; kt += 32) {
      bf16x16 a = frag_a(xs, KIN, kt, lane);
      acc0 = wmma_bf16(a, frag_b(wih_l, KIN, n0, kt, lane), acc0);
      acc1 = wmma_bf16(a, frag_b(wih_l, KIN, n0 + 16, kt, lane), acc1);
    }
#pragma unroll
    for (int kt = 0; kt < H; kt += 32) {
      bf16x16 a = frag_a(hs, H, kt, lane);
      acc0 = wmma_bf16(a, frag_b(whh_l, H, n0, kt, lane), acc0);
      acc1 = wmma_bf16(a, frag_b(whh_l, H, n0 + 16, kt, lane), acc1);
    }
    {
      const int N = lane & 15, mo = (lane & 16) ? 8 : 0;
#pragma unroll
      for (int r = 0; r < 8; ++r) {
        gl[(r + mo) * 256 + n0 + N]      = acc0[r];
        gl[(r + mo) * 256 + n0 + 16 + N] = acc1[r];
      }
    }
    __syncthreads();

    // Activations for this WG's 16x64 cells; gate order i,f,g,o.
    for (int i = tid; i < 16 * 64; i += 256) {
      int m = i >> 6, j = i & 63;
      float iv = sigmoidf_(gl[m * 256 + j]);
      float fv = sigmoidf_(gl[m * 256 + 64 + j]);
      float gv = tanhf(gl[m * 256 + 128 + j]);
      float ov = sigmoidf_(gl[m * 256 + 192 + j]);
      float c  = fv * cl[i] + iv * gv;
      cl[i] = c;
      float h = ov * tanhf(c);
      bf16_t hb = (bf16_t)h;
      int b = bt * 16 + m, hid = hsl * 64 + j;
      hbuf[(size_t)((t + 1) & 1) * (32 * H) + (size_t)b * H + hid] = hb;
      hout[((size_t)b * S + t) * H + hid] = hb;
    }
    __threadfence();
    __syncthreads();
    if (tid == 0) {
      __hip_atomic_fetch_add(bar + bt, 1u, __ATOMIC_RELEASE, __HIP_MEMORY_SCOPE_AGENT);
      const unsigned tgt = 4u * (unsigned)(t + 1);
      while (__hip_atomic_load(bar + bt, __ATOMIC_ACQUIRE, __HIP_MEMORY_SCOPE_AGENT) < tgt)
        __builtin_amdgcn_s_sleep(2);
    }
    __syncthreads();
  }
}

// ---------------------------------------------------------------------------
// Head 1: pred = silu(h1 @ W1^T + b1); pred += silu(pred @ W1b^T + b1b);
// write transposed to predT (B, 64, S) in bf16. One WG per (b, 64-row s chunk).
// ---------------------------------------------------------------------------
__global__ void head1_kernel(const bf16_t* __restrict__ h1,   // (B,S,256)
                             const bf16_t* __restrict__ W1,   // (64,256)
                             const float*  __restrict__ b1,
                             const bf16_t* __restrict__ W1b,  // (64,64)
                             const float*  __restrict__ b1b,
                             bf16_t* __restrict__ predT) {    // (B,64,S)
  constexpr int S = 2048;
  extern __shared__ char smem[];
  bf16_t* hst  = (bf16_t*)smem;            // 64 x 256
  bf16_t* w1l  = hst + 64 * 256;           // 64 x 256
  bf16_t* w1bl = w1l + 64 * 256;           // 64 x 64
  float*  pf   = (float*)(w1bl + 64 * 64); // 64 x 64  (silu'd pred, f32)
  bf16_t* pb   = (bf16_t*)(pf + 64 * 64);  // 64 x 64  (bf16 copy)

  const int tid = threadIdx.x, lane = tid & 31, wv = tid >> 5;
  const int b = blockIdx.x >> 5;
  const int s0 = (blockIdx.x & 31) * 64;

  for (int i = tid; i < 64 * 256 / 8; i += 256) {
    async_g2l_b128(smem, w1l + i * 8, W1 + i * 8);
    async_g2l_b128(smem, hst + i * 8, h1 + ((size_t)b * S + s0) * 256 + i * 8);
  }
  for (int i = tid; i < 64 * 64 / 8; i += 256)
    async_g2l_b128(smem, w1bl + i * 8, W1b + i * 8);
  wait_async0();
  __syncthreads();

  const int N = lane & 15, mo = (lane & 16) ? 8 : 0;
  // 16 tiles (4 M x 4 N), 2 per wave.
  for (int ti = wv * 2; ti < wv * 2 + 2; ++ti) {
    int mt = ti >> 2, nt = ti & 3;
    floatx8 acc = bcast8(b1[nt * 16 + N]);
#pragma unroll
    for (int kt = 0; kt < 256; kt += 32)
      acc = wmma_bf16(frag_a(hst + mt * 16 * 256, 256, kt, lane),
                      frag_b(w1l, 256, nt * 16, kt, lane), acc);
#pragma unroll
    for (int r = 0; r < 8; ++r) {
      float v = siluf_(acc[r]);
      int m = mt * 16 + r + mo, n = nt * 16 + N;
      pf[m * 64 + n] = v;
      pb[m * 64 + n] = (bf16_t)v;
    }
  }
  __syncthreads();
  for (int ti = wv * 2; ti < wv * 2 + 2; ++ti) {
    int mt = ti >> 2, nt = ti & 3;
    floatx8 acc = bcast8(b1b[nt * 16 + N]);
#pragma unroll
    for (int kt = 0; kt < 64; kt += 32)
      acc = wmma_bf16(frag_a(pb + mt * 16 * 64, 64, kt, lane),
                      frag_b(w1bl, 64, nt * 16, kt, lane), acc);
#pragma unroll
    for (int r = 0; r < 8; ++r) {
      int m = mt * 16 + r + mo, n = nt * 16 + N;
      float v = pf[m * 64 + n] + siluf_(acc[r]);
      predT[((size_t)b * 64 + n) * S + s0 + m] = (bf16_t)v;
    }
  }
}

// ---------------------------------------------------------------------------
// Head 2a: P2 = predT @ W2^T + b2  -> (B,64,2048) f32.
// One WG per (b, 128-wide os chunk). Full A (64x2048 bf16 = 256KB) async-loaded
// into LDS; B fragments stream from L2 with global_prefetch ahead.
// ---------------------------------------------------------------------------
__global__ void head2a_kernel(const bf16_t* __restrict__ predT, // (B,64,2048)
                              const bf16_t* __restrict__ W2,    // (2048,2048)
                              const float*  __restrict__ b2,
                              float* __restrict__ P2) {         // (B,64,2048)
  extern __shared__ char smem[];
  bf16_t* Al = (bf16_t*)smem;  // 64 x 2048
  const int tid = threadIdx.x, lane = tid & 31, wv = tid >> 5;
  const int b = blockIdx.x >> 4;
  const int os0 = (blockIdx.x & 15) * 128;

  for (int i = tid; i < 64 * 2048 / 8; i += 256)
    async_g2l_b128(smem, Al + (size_t)i * 8,
                   predT + (size_t)b * 64 * 2048 + (size_t)i * 8);
  wait_async0();
  __syncthreads();

  const int N = lane & 15, mo = (lane & 16) ? 8 : 0;
  const int mt = wv >> 1, nq = (wv & 1) * 4;  // wave: 1 M-tile x 4 N-tiles
  floatx8 acc[4];
#pragma unroll
  for (int j = 0; j < 4; ++j) acc[j] = bcast8(b2[os0 + (nq + j) * 16 + N]);
  for (int kt = 0; kt < 2048; kt += 32) {
    if (kt + 32 < 2048)
      __builtin_prefetch(W2 + (size_t)(os0 + nq * 16 + N) * 2048 + kt + 32, 0, 1);
    bf16x16 a = frag_a(Al + mt * 16 * 2048, 2048, kt, lane);
#pragma unroll
    for (int j = 0; j < 4; ++j)
      acc[j] = wmma_bf16(a, frag_b(W2, 2048, os0 + (nq + j) * 16, kt, lane), acc[j]);
  }
#pragma unroll
  for (int j = 0; j < 4; ++j) {
#pragma unroll
    for (int r = 0; r < 8; ++r) {
      int of = mt * 16 + r + mo, os = os0 + (nq + j) * 16 + N;
      P2[((size_t)b * 64 + of) * 2048 + os] = acc[j][r];
    }
  }
}

// ---------------------------------------------------------------------------
// Head 2b: P3 = P2 + silu(P2 @ W2b^T + b2b); out = softmax(P3, axis=-1).
// One WG per batch element; P2[b] as bf16 (256KB) in LDS; P3 in-place in P2.
// ---------------------------------------------------------------------------
__global__ void head2b_kernel(const float*  __restrict__ b2b,
                              const bf16_t* __restrict__ W2b,  // (2048,2048)
                              float* __restrict__ P2,          // in P2 / out P3
                              float* __restrict__ out) {       // (B,64,2048)
  extern __shared__ char smem[];
  bf16_t* Al = (bf16_t*)smem;  // 64 x 2048
  const int tid = threadIdx.x, lane = tid & 31, wv = tid >> 5;
  const int b = blockIdx.x;
  float* P2b = P2 + (size_t)b * 64 * 2048;

  for (int i = tid; i < 64 * 2048; i += 256) Al[i] = (bf16_t)P2b[i];
  __syncthreads();

  const int N = lane & 15, mo = (lane & 16) ? 8 : 0;
  const int mt = wv >> 1, nq = (wv & 1) * 4;
  for (int os0 = 0; os0 < 2048; os0 += 128) {
    floatx8 acc[4];
#pragma unroll
    for (int j = 0; j < 4; ++j) acc[j] = bcast8(b2b[os0 + (nq + j) * 16 + N]);
    for (int kt = 0; kt < 2048; kt += 32) {
      if (kt + 32 < 2048)
        __builtin_prefetch(W2b + (size_t)(os0 + nq * 16 + N) * 2048 + kt + 32, 0, 1);
      bf16x16 a = frag_a(Al + mt * 16 * 2048, 2048, kt, lane);
#pragma unroll
      for (int j = 0; j < 4; ++j)
        acc[j] = wmma_bf16(a, frag_b(W2b, 2048, os0 + (nq + j) * 16, kt, lane), acc[j]);
    }
#pragma unroll
    for (int j = 0; j < 4; ++j) {
#pragma unroll
      for (int r = 0; r < 8; ++r) {
        int of = mt * 16 + r + mo, os = os0 + (nq + j) * 16 + N;
        float p2v = (float)Al[of * 2048 + os];
        P2b[(size_t)of * 2048 + os] = p2v + siluf_(acc[j][r]);
      }
    }
  }
  __threadfence();
  __syncthreads();

  // Row-wise softmax over 2048; 8 rows per wave, wave32 shuffles.
  for (int row = wv; row < 64; row += 8) {
    const float* rp = P2b + (size_t)row * 2048;
    float* op = out + ((size_t)b * 64 + row) * 2048;
    float mx = -3.4e38f;
    for (int i = lane; i < 2048; i += 32) mx = fmaxf(mx, rp[i]);
#pragma unroll
    for (int off = 16; off > 0; off >>= 1) mx = fmaxf(mx, __shfl_xor(mx, off, 32));
    float sum = 0.f;
    for (int i = lane; i < 2048; i += 32) {
      float e = expf(rp[i] - mx);
      sum += e;
      op[i] = e;
    }
#pragma unroll
    for (int off = 16; off > 0; off >>= 1) sum += __shfl_xor(sum, off, 32);
    float inv = 1.0f / sum;
    for (int i = lane; i < 2048; i += 32) op[i] *= inv;
  }
}

// ---------------------------------------------------------------------------
// Host launcher
// ---------------------------------------------------------------------------
extern "C" void kernel_launch(void* const* d_in, const int* in_sizes, int n_in,
                              void* d_out, int out_size, void* d_ws, size_t ws_size,
                              hipStream_t stream) {
  (void)in_sizes; (void)n_in; (void)out_size; (void)ws_size;
  const float* x    = (const float*)d_in[0];
  const float* Wih0 = (const float*)d_in[1];
  const float* Whh0 = (const float*)d_in[2];
  const float* bih0 = (const float*)d_in[3];
  const float* bhh0 = (const float*)d_in[4];
  const float* Wih1 = (const float*)d_in[5];
  const float* Whh1 = (const float*)d_in[6];
  const float* bih1 = (const float*)d_in[7];
  const float* bhh1 = (const float*)d_in[8];
  const float* W1   = (const float*)d_in[9];
  const float* b1   = (const float*)d_in[10];
  const float* W1b  = (const float*)d_in[11];
  const float* b1b  = (const float*)d_in[12];
  const float* W2   = (const float*)d_in[13];
  const float* b2   = (const float*)d_in[14];
  const float* W2b  = (const float*)d_in[15];
  const float* b2b  = (const float*)d_in[16];

  char* w = (char*)d_ws;
  size_t off = 0;
  auto carve = [&](size_t bytes) -> char* {
    char* p = w + off;
    off = (off + bytes + 255) & ~(size_t)255;
    return p;
  };
  bf16_t* xt    = (bf16_t*)carve((size_t)32 * 2048 * 64 * 2);
  bf16_t* wih0b = (bf16_t*)carve((size_t)1024 * 64 * 2);
  bf16_t* whh0b = (bf16_t*)carve((size_t)1024 * 256 * 2);
  bf16_t* wih1b = (bf16_t*)carve((size_t)1024 * 256 * 2);
  bf16_t* whh1b = (bf16_t*)carve((size_t)1024 * 256 * 2);
  bf16_t* w1bf  = (bf16_t*)carve((size_t)64 * 256 * 2);
  bf16_t* w1bbf = (bf16_t*)carve((size_t)64 * 64 * 2);
  bf16_t* w2bf  = (bf16_t*)carve((size_t)2048 * 2048 * 2);
  bf16_t* w2bbf = (bf16_t*)carve((size_t)2048 * 2048 * 2);
  float*  bias0 = (float*)carve(1024 * 4);
  float*  bias1 = (float*)carve(1024 * 4);
  bf16_t* h0    = (bf16_t*)carve((size_t)32 * 2048 * 256 * 2);
  bf16_t* h1o   = (bf16_t*)carve((size_t)32 * 2048 * 256 * 2);
  bf16_t* predT = (bf16_t*)carve((size_t)32 * 64 * 2048 * 2);
  float*  P2    = (float*)carve((size_t)32 * 64 * 2048 * 4);
  char* zbase = w + off;  // contiguous zero-init zone
  bf16_t*   hbuf0 = (bf16_t*)carve((size_t)2 * 32 * 256 * 2);
  bf16_t*   hbuf1 = (bf16_t*)carve((size_t)2 * 32 * 256 * 2);
  unsigned* bar   = (unsigned*)carve(256);
  const int zwords = (int)(((w + off) - zbase) / 4);

  k_zero_u32<<<(zwords + 255) / 256, 256, 0, stream>>>((unsigned*)zbase, zwords);

  auto cvt = [&](const float* s, bf16_t* d, int n) {
    k_cvt_bf16<<<(n + 255) / 256, 256, 0, stream>>>(s, d, n);
  };
  cvt(Wih0, wih0b, 1024 * 64);
  cvt(Whh0, whh0b, 1024 * 256);
  cvt(Wih1, wih1b, 1024 * 256);
  cvt(Whh1, whh1b, 1024 * 256);
  cvt(W1, w1bf, 64 * 256);
  cvt(W1b, w1bbf, 64 * 64);
  cvt(W2, w2bf, 2048 * 2048);
  cvt(W2b, w2bbf, 2048 * 2048);
  k_bias_sum<<<4, 256, 0, stream>>>(bih0, bhh0, bias0, 1024);
  k_bias_sum<<<4, 256, 0, stream>>>(bih1, bhh1, bias1, 1024);
  k_xt<<<(32 * 2048 * 64 + 255) / 256, 256, 0, stream>>>(x, xt);

  // LSTM layer 0 (KIN=64): LDS = 196KB ; layer 1 (KIN=256): LDS = 292KB (<320KB/WGP)
  size_t sh0 = (size_t)(256 * 256 + 256 * 64 + 16 * 256 + 16 * 256) * 2 +
               (size_t)(16 * 256 + 16 * 64) * 4;
  lstm_layer_kernel<64><<<8, 256, sh0, stream>>>(xt, wih0b, whh0b, bias0, h0,
                                                 hbuf0, bar);
  size_t sh1 = (size_t)(256 * 256 + 256 * 256 + 16 * 256 + 16 * 256) * 2 +
               (size_t)(16 * 256 + 16 * 64) * 4;
  lstm_layer_kernel<256><<<8, 256, sh1, stream>>>(h0, wih1b, whh1b, bias1, h1o,
                                                  hbuf1, bar + 16);

  size_t shh1 = (size_t)(64 * 256 + 64 * 256 + 64 * 64) * 2 +
                (size_t)(64 * 64) * 4 + (size_t)(64 * 64) * 2;
  head1_kernel<<<32 * 32, 256, shh1, stream>>>(h1o, w1bf, b1, w1bbf, b1b, predT);

  head2a_kernel<<<32 * 16, 256, (size_t)64 * 2048 * 2, stream>>>(predT, w2bf, b2, P2);
  head2b_kernel<<<32, 256, (size_t)64 * 2048 * 2, stream>>>(b2b, w2bbf, P2,
                                                            (float*)d_out);
}
